// S2SPairBlock_4063039062764
// MI455X (gfx1250) — matrix operand order repaired and checked by reading, not compiled, not run on hardware
//
#include <hip/hip_runtime.h>
#include <hip/hip_bf16.h>
#include <math.h>

// ---------------------------------------------------------------------------
// S2S pair block on gfx1250 (MI455X). All heavy GEMMs use
// v_wmma_f32_16x16x32_f16 (f16 operands, fp32 accumulate). Weights are
// pre-swizzled into the WMMA B-operand fragment layout so each lane loads its
// 32-byte v16h fragment contiguously from global (L2-resident, ~200KB total,
// hoisted into VGPRs by the compiler). Activations are staged in LDS as f16
// rows; A-fragments are gathered with packed 32-bit LDS reads per the CDNA5
// 16-bit A-matrix layout (compiler fuses them into ds_load_b128 pairs).
// All transcendental paths (gelu/unit/LN) use branchless v_exp/v_rcp/v_rsq/
// v_sqrt so no EXEC-divergent libm expansions sit between WMMA stages.
// ---------------------------------------------------------------------------

typedef __attribute__((ext_vector_type(16))) _Float16 v16h;
typedef __attribute__((ext_vector_type(8)))  float    v8f;

#define NN 1024
#define KK 32
#define AA 5
#define PP 64
#define DD 256

// Branchless tanh-approx gelu: tanh(u) = 1 - 2/(1+exp(2u)).
// exp overflow -> inf -> rcp -> 0 -> tanh=+1 ; exp underflow -> 0 -> tanh=-1.
__device__ __forceinline__ float gelu_f(float x){
  float u = 0.7978845608028654f*(x + 0.044715f*x*x*x);
  float e = __expf(2.0f*u);
  float t = 1.0f - 2.0f*__builtin_amdgcn_rcpf(e + 1.0f);
  return 0.5f*x*(1.0f + t);
}
__device__ __forceinline__ float softplus_f(float x){
  return (x > 20.f) ? x : log1pf(expf(x));
}
__device__ __forceinline__ float rcp_f(float x){ return __builtin_amdgcn_rcpf(x); }
__device__ __forceinline__ float rsq_f(float x){ return __builtin_amdgcn_rsqf(x); }
__device__ __forceinline__ float sqrt_f(float x){ return __builtin_amdgcn_sqrtf(x); }

// A-fragment (16x32 f16) from row-major LDS tile. ISA layout:
// lane m=lane&15, half=lane>>4; vgpr j<4 holds K = 8*half+2j(+1),
// vgpr j>=4 holds K = 16+8*half+2(j-4)(+1)  -> packed dword gathers
// (lowered to two ds_load_b128 at byte offsets 16*half and 32+16*half).
__device__ __forceinline__ v16h load_a_frag(const _Float16* Abase, int lda, int m, int half){
  union { v16h h; unsigned u[8]; } r;
  const unsigned* row = reinterpret_cast<const unsigned*>(Abase + m*lda);
  int b = half << 2;
  r.u[0] = row[b+0]; r.u[1] = row[b+1]; r.u[2] = row[b+2]; r.u[3] = row[b+3];
  r.u[4] = row[b+8]; r.u[5] = row[b+9]; r.u[6] = row[b+10]; r.u[7] = row[b+11];
  return r.h;
}

// B-fragment: pre-swizzled in global so lane's 16 halves are contiguous.
__device__ __forceinline__ v16h load_b_frag(const _Float16* w, int ntile, int ks, int nK, int lane){
  return *reinterpret_cast<const v16h*>(w + ((size_t)(((ntile*nK + ks) << 5) + lane) << 4));
}

__device__ __forceinline__ v8f wmma_f16(v16h a, v16h b, v8f c){
  return __builtin_amdgcn_wmma_f32_16x16x32_f16(false, a, false, b, (short)0, c, false, false);
}

// ---------------------------------------------------------------------------
// Weight prep: f32 -> f16, swizzled to B-fragment layout.
// flat = ((ntile*nK + ks)*32 + lane)*16 + e ; lane: n=lane&15, half=lane>>4;
// e: source = B[32*ks + 8*half + (e&7) + 16*(e>>3)][16*ntile + n]
// Concat columns: col < csplit -> w1, else w2. Rows >= realK are zero (K pad).
// ---------------------------------------------------------------------------
__global__ __launch_bounds__(256) void prep_weights(
    const float* __restrict__ w1, const float* __restrict__ w2,
    _Float16* __restrict__ dst, int realK, int ncols, int nK, int csplit, int total)
{
  int f = blockIdx.x*blockDim.x + threadIdx.x;
  if (f >= total) return;
  int e    = f & 15;
  int lane = (f >> 4) & 31;
  int rest = f >> 9;
  int ks = rest % nK;
  int nt = rest / nK;
  int n = lane & 15, half = lane >> 4, j = e >> 1, lo = e & 1;
  int sk = 32*ks + 8*half + (e & 7) + ((e >> 3) << 4);
  int sc = 16*nt + n;
  float v = 0.f;
  if (sk < realK){
    v = (sc < csplit) ? w1[sk*csplit + sc] : w2[sk*(ncols - csplit) + (sc - csplit)];
  }
  dst[f] = (_Float16)v;
}

// ---------------------------------------------------------------------------
// K0: per-node: LN(features), ol=gelu(local@w_ol), lor=local@w_or,
// al=local@w_al, ar=local@w_ar, and rigid frame (R, t=ca).
// ---------------------------------------------------------------------------
__global__ __launch_bounds__(256) void k0_node_prep(
    const float* __restrict__ features, const float* __restrict__ lnl_s, const float* __restrict__ lnl_b,
    const float* __restrict__ w_ol, const float* __restrict__ w_or,
    const float* __restrict__ w_al, const float* __restrict__ w_ar,
    const float* __restrict__ pos,
    float* __restrict__ g_ol, float* __restrict__ g_lor,
    float* __restrict__ g_al, float* __restrict__ g_ar,
    float* __restrict__ g_R, float* __restrict__ g_t)
{
  __shared__ float sBuf[256];
  __shared__ float sLoc[256];
  int n = blockIdx.x, t = threadIdx.x;
  float f = features[n*DD + t];
  sBuf[t] = f; __syncthreads();
  for (int s = 128; s > 0; s >>= 1){ if (t < s) sBuf[t] += sBuf[t+s]; __syncthreads(); }
  float m = sBuf[0] * (1.0f/DD);
  __syncthreads();
  float d = f - m;
  sBuf[t] = d*d; __syncthreads();
  for (int s = 128; s > 0; s >>= 1){ if (t < s) sBuf[t] += sBuf[t+s]; __syncthreads(); }
  float rstd = rsq_f(sBuf[0]*(1.0f/DD) + 1e-5f);
  sLoc[t] = d*rstd*lnl_s[t] + lnl_b[t];
  __syncthreads();
  if (t < 144){
    float acc = 0.f;
    if (t < 8){
      for (int c = 0; c < DD; ++c) acc += sLoc[c]*w_ol[c*8 + t];
      g_ol[n*8 + t] = gelu_f(acc);
    } else if (t < 16){
      int j = t - 8;
      for (int c = 0; c < DD; ++c) acc += sLoc[c]*w_or[c*8 + j];
      g_lor[n*8 + j] = acc;
    } else if (t < 80){
      int j = t - 16;
      for (int c = 0; c < DD; ++c) acc += sLoc[c]*w_al[c*PP + j];
      g_al[n*PP + j] = acc;
    } else {
      int j = t - 80;
      for (int c = 0; c < DD; ++c) acc += sLoc[c]*w_ar[c*PP + j];
      g_ar[n*PP + j] = acc;
    }
  }
  if (t == 0){
    float Na[3], Ca[3], Cc[3];
    for (int j = 0; j < 3; ++j){
      Na[j] = pos[n*AA*3 + j];
      Ca[j] = pos[n*AA*3 + 3 + j];
      Cc[j] = pos[n*AA*3 + 6 + j];
    }
    float e1[3], e2[3], e3[3], u[3];
    float nrm = 0.f;
    for (int j = 0; j < 3; ++j){ e1[j] = Cc[j]-Ca[j]; nrm += e1[j]*e1[j]; }
    float inv = rcp_f(fmaxf(sqrt_f(nrm), 1e-8f));
    for (int j = 0; j < 3; ++j) e1[j] *= inv;
    float du = 0.f;
    for (int j = 0; j < 3; ++j){ u[j] = Na[j]-Ca[j]; du += u[j]*e1[j]; }
    nrm = 0.f;
    for (int j = 0; j < 3; ++j){ e2[j] = u[j]-du*e1[j]; nrm += e2[j]*e2[j]; }
    inv = rcp_f(fmaxf(sqrt_f(nrm), 1e-8f));
    for (int j = 0; j < 3; ++j) e2[j] *= inv;
    e3[0] = e1[1]*e2[2]-e1[2]*e2[1];
    e3[1] = e1[2]*e2[0]-e1[0]*e2[2];
    e3[2] = e1[0]*e2[1]-e1[1]*e2[0];
    for (int j = 0; j < 3; ++j){
      g_R[n*9 + j*3 + 0] = e1[j];
      g_R[n*9 + j*3 + 1] = e2[j];
      g_R[n*9 + j*3 + 2] = e3[j];
      g_t[n*3 + j] = Ca[j];
    }
  }
}

// ---------------------------------------------------------------------------
// Stage 1: one WG per node, rows = k 0..31.
// feat[192] = [LN(pair) | ol x lor[nb] outer | al + ar[nb]]  (LDS f16)
// concat GEMM -> gelu-gate -> hidden[128] -> @w_o1 -> + pair = pairmid
// ---------------------------------------------------------------------------
__global__ __launch_bounds__(256) void k_stage1(
    const float* __restrict__ pair, const int* __restrict__ nbr,
    const float* __restrict__ ln1_s, const float* __restrict__ ln1_b,
    const float* __restrict__ g_ol, const float* __restrict__ g_lor,
    const float* __restrict__ g_al, const float* __restrict__ g_ar,
    const _Float16* __restrict__ wG1H1, const _Float16* __restrict__ wO1,
    float* __restrict__ g_pm)
{
  __shared__ _Float16 sA[32*192];
  __shared__ _Float16 sH[32*128];
  __shared__ float sMean[32], sRstd[32];
  __shared__ int sNb[32];
  int n = blockIdx.x, tid = threadIdx.x;
  if (tid < 32){
    sNb[tid] = nbr[n*KK + tid];
    const float* pr = pair + (size_t)(n*KK + tid)*PP;
    float s = 0.f;
    for (int c = 0; c < PP; ++c) s += pr[c];
    float mm = s * (1.0f/PP);
    float ss = 0.f;
    for (int c = 0; c < PP; ++c){ float dd = pr[c]-mm; ss += dd*dd; }
    sMean[tid] = mm;
    sRstd[tid] = rsq_f(ss*(1.0f/PP) + 1e-5f);
  }
  __syncthreads();
  for (int idx = tid; idx < 32*192; idx += 256){
    int row = idx / 192, c = idx - row*192;
    int nb = sNb[row];
    float v;
    if (c < 64){
      v = (pair[(size_t)(n*KK+row)*PP + c] - sMean[row]) * sRstd[row] * ln1_s[c] + ln1_b[c];
    } else if (c < 128){
      int q = c - 64;
      v = g_ol[n*8 + (q>>3)] * g_lor[nb*8 + (q&7)];
    } else {
      int q = c - 128;
      v = g_al[n*PP + q] + g_ar[nb*PP + q];
    }
    sA[idx] = (_Float16)v;
  }
  __syncthreads();
  int wave = tid >> 5, lane = tid & 31;
  int nn = lane & 15, half = lane >> 4;
  // wave w owns g1 col-tile w and h1 col-tile 8+w (concat cols 256)
  for (int rt = 0; rt < 2; ++rt){
    v8f accG = {}; v8f accH = {};
    for (int ks = 0; ks < 6; ++ks){
      v16h a  = load_a_frag(sA + (rt*16)*192 + ks*32, 192, nn, half);
      v16h bg = load_b_frag(wG1H1, wave,     ks, 6, lane);
      v16h bh = load_b_frag(wG1H1, 8 + wave, ks, 6, lane);
      accG = wmma_f16(a, bg, accG);
      accH = wmma_f16(a, bh, accH);
    }
    for (int r = 0; r < 8; ++r){
      int row = rt*16 + r + 8*half;
      sH[row*128 + wave*16 + nn] = (_Float16)(gelu_f(accG[r]) * accH[r]);
    }
  }
  __syncthreads();
  {
    int ct = wave & 3, rt = wave >> 2;
    v8f acc = {};
    for (int ks = 0; ks < 4; ++ks){
      v16h a = load_a_frag(sH + (rt*16)*128 + ks*32, 128, nn, half);
      v16h b = load_b_frag(wO1, ct, ks, 4, lane);
      acc = wmma_f16(a, b, acc);
    }
    for (int r = 0; r < 8; ++r){
      int row = rt*16 + r + 8*half;
      size_t gi = (size_t)(n*KK + row)*PP + ct*16 + nn;
      g_pm[gi] = acc[r] + pair[gi];
    }
  }
}

// ---------------------------------------------------------------------------
// K1: per (n,k): p2 = LN(pairmid), npos = R^T(pos[nb]-t), pf[96] f16 row
// (pf = [p2(64) | dscale*npos(15) | unit(npos)(15) | 0 0]); store npos f32.
// ---------------------------------------------------------------------------
__global__ __launch_bounds__(64) void k1_pf(
    const float* __restrict__ g_pm, const float* __restrict__ ln2_s, const float* __restrict__ ln2_b,
    const int* __restrict__ nbr, const float* __restrict__ pos,
    const float* __restrict__ g_R, const float* __restrict__ g_t,
    const float* __restrict__ d_scale,
    _Float16* __restrict__ g_pf, float* __restrict__ g_np)
{
  __shared__ float sRed[64];
  __shared__ float sNp[15];
  int idx = blockIdx.x;           // n*K + k
  int n = idx >> 5;
  int t = threadIdx.x;
  float v = g_pm[(size_t)idx*PP + t];
  sRed[t] = v; __syncthreads();
  for (int s = 32; s > 0; s >>= 1){ if (t < s) sRed[t] += sRed[t+s]; __syncthreads(); }
  float m = sRed[0] * (1.0f/PP);
  __syncthreads();
  float d = v - m;
  sRed[t] = d*d; __syncthreads();
  for (int s = 32; s > 0; s >>= 1){ if (t < s) sRed[t] += sRed[t+s]; __syncthreads(); }
  float rstd = rsq_f(sRed[0]*(1.0f/PP) + 1e-5f);
  _Float16* pf = g_pf + (size_t)idx*96;
  pf[t] = (_Float16)(d*rstd*ln2_s[t] + ln2_b[t]);
  if (t < 15){
    int nb = nbr[idx];
    int a = t/3, i = t - a*3;
    float acc = 0.f;
    for (int j = 0; j < 3; ++j){
      float xc = pos[nb*AA*3 + a*3 + j] - g_t[n*3 + j];
      acc += xc * g_R[n*9 + j*3 + i];
    }
    sNp[t] = acc;
    g_np[(size_t)idx*15 + t] = acc;
  }
  __syncthreads();
  if (t < 15){
    int a = t/3;
    float nrm = sqrt_f(sNp[a*3]*sNp[a*3] + sNp[a*3+1]*sNp[a*3+1] + sNp[a*3+2]*sNp[a*3+2]);
    float ds = softplus_f(d_scale[0]) * 0.1f;
    pf[64 + t] = (_Float16)(ds * sNp[t]);
    pf[79 + t] = (_Float16)(sNp[t] * rcp_f(fmaxf(nrm, 1e-8f)));
  }
  if (t == 0){ pf[94] = (_Float16)0.f; pf[95] = (_Float16)0.f; }
}

// ---------------------------------------------------------------------------
// K2: left/right = pf @ [w_l | w_r]  (one WG per node, 32 rows, K=96)
// ---------------------------------------------------------------------------
__global__ __launch_bounds__(256) void k2_leftright(
    const _Float16* __restrict__ g_pf, const _Float16* __restrict__ wLR,
    float* __restrict__ g_left, float* __restrict__ g_right)
{
  __shared__ _Float16 sPf[32*96];
  int n = blockIdx.x, tid = threadIdx.x;
  const unsigned* src = reinterpret_cast<const unsigned*>(g_pf + (size_t)n*KK*96);
  unsigned* dst = reinterpret_cast<unsigned*>(sPf);
  for (int i = tid; i < 32*96/2; i += 256) dst[i] = src[i];
  __syncthreads();
  int wave = tid >> 5, lane = tid & 31;
  int nn = lane & 15, half = lane >> 4;
  for (int rt = 0; rt < 2; ++rt){
    v8f acc = {};
    for (int ks = 0; ks < 3; ++ks){
      v16h a = load_a_frag(sPf + (rt*16)*96 + ks*32, 96, nn, half);
      v16h b = load_b_frag(wLR, wave, ks, 3, lane);
      acc = wmma_f16(a, b, acc);
    }
    for (int r = 0; r < 8; ++r){
      int row = rt*16 + r + 8*half;
      int col = wave*16 + nn;
      if (col < 64) g_left [(size_t)(n*KK + row)*PP + col]        = acc[r];
      else          g_right[(size_t)(n*KK + row)*PP + (col - 64)] = acc[r];
    }
  }
}

// ---------------------------------------------------------------------------
// Stage 2: one WG per (node, 4 k's): 32 rows = 4k x 8k2.
// ppf[224] = [left+right(64) | dscale*rel(75) | unit(rel)(75) | pad0(10)]
// -> gelu-gated GEMM (K=224) -> @w_o2 -> masked mean over k2 -> + pairmid
// ---------------------------------------------------------------------------
__global__ __launch_bounds__(128) void k_stage2(
    const float* __restrict__ g_pm, const float* __restrict__ g_left, const float* __restrict__ g_right,
    const float* __restrict__ g_np, const int* __restrict__ nbr, const unsigned char* __restrict__ mask,
    const float* __restrict__ d_scale,
    const _Float16* __restrict__ wG2H2, const _Float16* __restrict__ wO2,
    float* __restrict__ out)
{
  __shared__ _Float16 sP[32*224];
  __shared__ _Float16 sH2[32*64];
  __shared__ float sProj[32*64];
  __shared__ float sL[4*64];
  __shared__ float sR[8*64];
  __shared__ float sNpK[4*15];
  __shared__ float sNpK2[8*15];
  __shared__ int sPmK[4];
  __shared__ int sPmK2[8];
  int wg = blockIdx.x;
  int node = wg >> 3;
  int kb = (wg & 7) << 2;
  int tid = threadIdx.x;
  float ds = softplus_f(d_scale[0]) * 0.1f;
  for (int i = tid; i < 256; i += 128) sL[i]    = g_left [(size_t)(node*KK + kb + (i>>6))*PP + (i&63)];
  for (int i = tid; i < 512; i += 128) sR[i]    = g_right[(size_t)(node*KK + (i>>6))*PP + (i&63)];
  for (int i = tid; i < 60;  i += 128) sNpK[i]  = g_np[(size_t)(node*KK + kb + i/15)*15 + i%15];
  for (int i = tid; i < 120; i += 128) sNpK2[i] = g_np[(size_t)(node*KK + i/15)*15 + i%15];
  if (tid < 12){
    int k = (tid < 4) ? (kb + tid) : (tid - 4);
    int nb = nbr[node*KK + k];
    int pm = (mask[node] != 0) && (nb != -1) && (mask[nb] != 0);
    if (tid < 4) sPmK[tid] = pm; else sPmK2[tid-4] = pm;
  }
  __syncthreads();
  for (int idx = tid; idx < 32*224; idx += 128){
    int row = idx / 224, c = idx - row*224;
    int kl = row >> 3, k2 = row & 7;
    float v = 0.f;
    if (c < 64){
      v = sL[kl*64 + c] + sR[k2*64 + c];
    } else if (c < 214){
      int q = c - 64;
      int isdir = (q >= 75) ? 1 : 0;
      if (isdir) q -= 75;
      int p = q/3, i = q - p*3;
      int a = p/5, b = p - a*5;
      float d0 = sNpK[kl*15 + a*3 + 0] - sNpK2[k2*15 + b*3 + 0];
      float d1 = sNpK[kl*15 + a*3 + 1] - sNpK2[k2*15 + b*3 + 1];
      float d2 = sNpK[kl*15 + a*3 + 2] - sNpK2[k2*15 + b*3 + 2];
      float di = (i==0) ? d0 : ((i==1) ? d1 : d2);
      if (isdir){
        float nrm = sqrt_f(d0*d0 + d1*d1 + d2*d2);
        v = di * rcp_f(fmaxf(nrm, 1e-8f));
      } else {
        v = ds * di;
      }
    }
    sP[idx] = (_Float16)v;
  }
  __syncthreads();
  int wave = tid >> 5, lane = tid & 31;
  int nn = lane & 15, half = lane >> 4;
  // wave w owns g2 col-tile w and h2 col-tile 4+w (concat cols 128)
  for (int rt = 0; rt < 2; ++rt){
    v8f accG = {}; v8f accH = {};
    for (int ks = 0; ks < 7; ++ks){
      v16h a  = load_a_frag(sP + (rt*16)*224 + ks*32, 224, nn, half);
      v16h bg = load_b_frag(wG2H2, wave,     ks, 7, lane);
      v16h bh = load_b_frag(wG2H2, 4 + wave, ks, 7, lane);
      accG = wmma_f16(a, bg, accG);
      accH = wmma_f16(a, bh, accH);
    }
    for (int r = 0; r < 8; ++r){
      int row = rt*16 + r + 8*half;
      sH2[row*64 + wave*16 + nn] = (_Float16)(gelu_f(accG[r]) * accH[r]);
    }
  }
  __syncthreads();
  for (int rt = 0; rt < 2; ++rt){
    v8f acc = {};
    for (int ks = 0; ks < 2; ++ks){
      v16h a = load_a_frag(sH2 + (rt*16)*64 + ks*32, 64, nn, half);
      v16h b = load_b_frag(wO2, wave, ks, 2, lane);
      acc = wmma_f16(a, b, acc);
    }
    for (int r = 0; r < 8; ++r){
      int row = rt*16 + r + 8*half;
      sProj[row*64 + wave*16 + nn] = acc[r];
    }
  }
  __syncthreads();
  for (int i = tid; i < 256; i += 128){
    int kl = i >> 6, c = i & 63;
    float num = 0.f; int den = 0;
    int pmk = sPmK[kl];
    for (int k2 = 0; k2 < 8; ++k2){
      if (pmk && sPmK2[k2]){ num += sProj[(kl*8 + k2)*64 + c]; den++; }
    }
    float res = num * rcp_f(den > 0 ? (float)den : 1.0f);
    size_t gi = (size_t)(node*KK + kb + kl)*PP + c;
    out[gi] = g_pm[gi] + res;
  }
}

// ---------------------------------------------------------------------------
extern "C" void kernel_launch(void* const* d_in, const int* in_sizes, int n_in,
                              void* d_out, int out_size, void* d_ws, size_t ws_size,
                              hipStream_t stream)
{
  (void)in_sizes; (void)n_in; (void)out_size; (void)ws_size;
  const float* features = (const float*)d_in[0];
  const float* pair     = (const float*)d_in[1];
  const float* pos      = (const float*)d_in[2];
  const int*   nbr      = (const int*)d_in[3];
  const unsigned char* mask = (const unsigned char*)d_in[4];
  const float* lnl_s = (const float*)d_in[5];
  const float* lnl_b = (const float*)d_in[6];
  const float* ln1_s = (const float*)d_in[7];
  const float* ln1_b = (const float*)d_in[8];
  const float* ln2_s = (const float*)d_in[9];
  const float* ln2_b = (const float*)d_in[10];
  const float* w_ol  = (const float*)d_in[11];
  const float* w_or  = (const float*)d_in[12];
  const float* w_al  = (const float*)d_in[13];
  const float* w_ar  = (const float*)d_in[14];
  const float* w_g1  = (const float*)d_in[15];
  const float* w_h1  = (const float*)d_in[16];
  const float* w_o1  = (const float*)d_in[17];
  const float* dscal = (const float*)d_in[18];
  const float* w_l   = (const float*)d_in[19];
  const float* w_r   = (const float*)d_in[20];
  const float* w_g2  = (const float*)d_in[21];
  const float* w_h2  = (const float*)d_in[22];
  const float* w_o2  = (const float*)d_in[23];
  float* out = (float*)d_out;

  char* base = (char*)d_ws;
  size_t off = 0;
  auto alloc = [&](size_t bytes)->char*{
    char* p = base + off;
    off = (off + bytes + 255) & ~(size_t)255;
    return p;
  };
  _Float16* WG1H1 = (_Float16*)alloc((size_t)49152*2);   // 192x256 swizzled
  _Float16* WO1   = (_Float16*)alloc((size_t)8192*2);    // 128x64
  _Float16* WLR   = (_Float16*)alloc((size_t)12288*2);   // 96x128 (94 real K)
  _Float16* WG2H2 = (_Float16*)alloc((size_t)28672*2);   // 224x128 (214 real K)
  _Float16* WO2   = (_Float16*)alloc((size_t)4096*2);    // 64x64
  float* g_ol  = (float*)alloc((size_t)NN*8*4);
  float* g_lor = (float*)alloc((size_t)NN*8*4);
  float* g_al  = (float*)alloc((size_t)NN*PP*4);
  float* g_ar  = (float*)alloc((size_t)NN*PP*4);
  float* g_R   = (float*)alloc((size_t)NN*9*4);
  float* g_t   = (float*)alloc((size_t)NN*3*4);
  float* g_pm  = (float*)alloc((size_t)NN*KK*PP*4);
  float* g_np  = (float*)alloc((size_t)NN*KK*15*4);
  _Float16* g_pf = (_Float16*)alloc((size_t)NN*KK*96*2);
  float* g_left  = (float*)alloc((size_t)NN*KK*PP*4);
  float* g_right = (float*)alloc((size_t)NN*KK*PP*4);

  // Weight prep (5 matrices)
  prep_weights<<<(49152+255)/256, 256, 0, stream>>>(w_g1, w_h1, WG1H1, 192, 256, 6, 128, 49152);
  prep_weights<<<(8192 +255)/256, 256, 0, stream>>>(w_o1, w_o1, WO1,   128,  64, 4,  64,  8192);
  prep_weights<<<(12288+255)/256, 256, 0, stream>>>(w_l,  w_r,  WLR,    94, 128, 3,  64, 12288);
  prep_weights<<<(28672+255)/256, 256, 0, stream>>>(w_g2, w_h2, WG2H2, 214, 128, 7,  64, 28672);
  prep_weights<<<(4096 +255)/256, 256, 0, stream>>>(w_o2, w_o2, WO2,    64,  64, 2,  64,  4096);

  k0_node_prep<<<NN, 256, 0, stream>>>(features, lnl_s, lnl_b, w_ol, w_or, w_al, w_ar, pos,
                                       g_ol, g_lor, g_al, g_ar, g_R, g_t);

  k_stage1<<<NN, 256, 0, stream>>>(pair, nbr, ln1_s, ln1_b, g_ol, g_lor, g_al, g_ar,
                                   WG1H1, WO1, g_pm);

  k1_pf<<<NN*KK, 64, 0, stream>>>(g_pm, ln2_s, ln2_b, nbr, pos, g_R, g_t, dscal, g_pf, g_np);

  k2_leftright<<<NN, 256, 0, stream>>>(g_pf, WLR, g_left, g_right);

  k_stage2<<<NN*8, 128, 0, stream>>>(g_pm, g_left, g_right, g_np, nbr, mask, dscal,
                                     WG2H2, WO2, out);
}
